// SelfAttention_53704271069444
// MI455X (gfx1250) — matrix operand order repaired and checked by reading, not compiled, not run on hardware
//
#include <hip/hip_runtime.h>

typedef __bf16 bf16;
typedef __attribute__((ext_vector_type(16))) __bf16 v16bf;
typedef __attribute__((ext_vector_type(8)))  __bf16 bf16x8;
typedef __attribute__((ext_vector_type(4)))  __bf16 bf16x4;
typedef __attribute__((ext_vector_type(8)))  float   v8f;
typedef __attribute__((ext_vector_type(4)))  int     v4i;

#define DEVI __device__ __forceinline__

constexpr int BATCH = 2;
constexpr int SEQ   = 2048;
constexpr int DIM   = 2048;
constexpr int HDIM  = 128;
constexpr int MROWS = BATCH * SEQ;               // 4096
constexpr float SM_SCALE = 0.08838834764831845f; // 1/sqrt(128)

// ---------------------------------------------------------------------------
// CDNA5 async global->LDS copy (ASYNCcnt path), with synchronous fallback.
// Builtin signature (from compiler diagnostic): (v4i AS1*, v4i AS3*, Ii, Ii).
// ---------------------------------------------------------------------------
#if defined(__gfx1250__) && __has_builtin(__builtin_amdgcn_global_load_async_to_lds_b128)
#define HAS_ASYNC_LDS 1
#else
#define HAS_ASYNC_LDS 0
#endif

DEVI void async_copy_b128(const bf16* gsrc, bf16* ldst) {
#if HAS_ASYNC_LDS
  __builtin_amdgcn_global_load_async_to_lds_b128(
      (__attribute__((address_space(1))) v4i*)(void*)gsrc,
      (__attribute__((address_space(3))) v4i*)(void*)ldst, 0, 0);
#else
  *reinterpret_cast<bf16x8*>(ldst) = *reinterpret_cast<const bf16x8*>(gsrc);
#endif
}

template <int N>
DEVI void wait_asynccnt() {
#if defined(__gfx1250__)
#if HAS_ASYNC_LDS
#if __has_builtin(__builtin_amdgcn_s_wait_asynccnt)
  __builtin_amdgcn_s_wait_asynccnt((short)N);
#else
  asm volatile("s_wait_asynccnt %0" ::"i"(N) : "memory");
#endif
#endif
#endif
}

// ---------------------------------------------------------------------------
// WMMA fragment helpers (layouts per CDNA5 ISA §7.12.2, bf16 K=32)
// ---------------------------------------------------------------------------

// A operand, 16x32 (MxK), from row-major [16 rows, ld cols] bf16.
// lane: m = lane&15, kb = (lane>>4)*8; elems 0..7 = K kb..kb+7, 8..15 = 16+kb..
DEVI v16bf load_a(const bf16* p, int ld) {
  const int lane = threadIdx.x & 31;
  const bf16* q = p + (size_t)(lane & 15) * ld + ((lane >> 4) << 3);
  bf16x8 lo = *reinterpret_cast<const bf16x8*>(q);
  bf16x8 hi = *reinterpret_cast<const bf16x8*>(q + 16);
  v16bf r;
#pragma unroll
  for (int i = 0; i < 8; ++i) { r[i] = lo[i]; r[i + 8] = hi[i]; }
  return r;
}

// B operand, 32x16 (KxN), supplied as row-major B^T [16 rows(N), ld cols(K)].
// lane: n = lane&15, kb = (lane>>4)*8; elems 0..7 = K kb..kb+7, 8..15 = 16+kb..
DEVI v16bf load_b(const bf16* p, int ld) {
  const int lane = threadIdx.x & 31;
  const bf16* q = p + (size_t)(lane & 15) * ld + ((lane >> 4) << 3);
  bf16x8 lo = *reinterpret_cast<const bf16x8*>(q);
  bf16x8 hi = *reinterpret_cast<const bf16x8*>(q + 16);
  v16bf r;
#pragma unroll
  for (int i = 0; i < 8; ++i) { r[i] = lo[i]; r[i + 8] = hi[i]; }
  return r;
}

DEVI v8f wmma_bf16(v16bf a, v16bf b, v8f c) {
  return __builtin_amdgcn_wmma_f32_16x16x32_bf16(false, a, false, b, (short)0, c,
                                                 false, false);
}

DEVI float redmax16(float v) {
#pragma unroll
  for (int m = 1; m < 16; m <<= 1) v = fmaxf(v, __shfl_xor(v, m, 32));
  return v;
}
DEVI float redsum16(float v) {
#pragma unroll
  for (int m = 1; m < 16; m <<= 1) v += __shfl_xor(v, m, 32);
  return v;
}

// ---------------------------------------------------------------------------
// Prep kernels
// ---------------------------------------------------------------------------

__global__ __launch_bounds__(256) void cvt_f32_bf16(const float* __restrict__ src,
                                                    bf16* __restrict__ dst, int n) {
  int i = (blockIdx.x * 256 + threadIdx.x) * 4;
  if (i + 3 < n) {
    float4 v = *reinterpret_cast<const float4*>(src + i);
    bf16x4 o;
    o[0] = (bf16)v.x; o[1] = (bf16)v.y; o[2] = (bf16)v.z; o[3] = (bf16)v.w;
    *reinterpret_cast<bf16x4*>(dst + i) = o;
  }
}

// Wt[n][k] = (bf16) W[k][n], dim x dim, LDS-tiled.
__global__ __launch_bounds__(256) void transpose_f32_bf16(const float* __restrict__ W,
                                                          bf16* __restrict__ Wt, int dim) {
  __shared__ float tile[32][33];
  const int tx = threadIdx.x & 31;
  const int ty = threadIdx.x >> 5;  // 0..7
  const int c0 = blockIdx.x * 32;
  const int r0 = blockIdx.y * 32;
#pragma unroll
  for (int i = 0; i < 32; i += 8)
    tile[ty + i][tx] = W[(size_t)(r0 + ty + i) * dim + c0 + tx];
  __syncthreads();
#pragma unroll
  for (int i = 0; i < 32; i += 8)
    Wt[(size_t)(c0 + ty + i) * dim + r0 + tx] = (bf16)tile[tx][ty + i];
}

// ---------------------------------------------------------------------------
// GEMM: C[M,N] = A[M,K](bf16) * Wt[N,K]^T(bf16) + bias, OutT epilogue.
// Block = 8 waves; wave w computes rows [by*128+16w, +16), cols [bx*128, +128).
// B tile (128 N x 32 K) is shared by all 8 waves -> staged into LDS via
// double-buffered async global->LDS DMA (overlaps previous tile's WMMAs).
// ---------------------------------------------------------------------------
template <typename OutT>
__global__ __launch_bounds__(256) void gemm_k(const bf16* __restrict__ A,
                                              const bf16* __restrict__ Wt,
                                              const float* __restrict__ bias,
                                              OutT* __restrict__ C,
                                              int Ndim, int Kdim) {
  // 40-elem rows: 80B stride = 16B-aligned, 20-dword bank period (gcd(20,64)=4,
  // 5 coprime 16 -> 16 distinct bank groups across the half-wave).
  __shared__ alignas(16) bf16 Bs[2][128][40];

  const int wave = threadIdx.x >> 5;
  const int row0 = blockIdx.y * 128 + wave * 16;
  const int col0 = blockIdx.x * 128;
  const int tid = threadIdx.x;

  auto stage = [&](int k0, int buf) {
#pragma unroll
    for (int i = 0; i < 2; ++i) {
      int idx = tid + i * 256;  // 0..511 : 128 rows x 4 chunks of 16B
      int n = idx >> 2;
      int c = (idx & 3) * 8;
      async_copy_b128(Wt + (size_t)(col0 + n) * Kdim + k0 + c, &Bs[buf][n][c]);
    }
  };

  v8f acc[8] = {};
  const bf16* Ap = A + (size_t)row0 * Kdim;
  stage(0, 0);
  for (int k0 = 0; k0 < Kdim; k0 += 32) {
    const int it = k0 >> 5;
    const bool more = (k0 + 32) < Kdim;
    if (more) stage(k0 + 32, (it + 1) & 1);
    if (more) wait_asynccnt<2>(); else wait_asynccnt<0>();  // own tile-i writes landed
    __syncthreads();                                        // everyone's writes visible
    __builtin_prefetch(Ap + k0 + 512, 0, 1);                // global_prefetch_b8
    v16bf a = load_a(Ap + k0, Kdim);
#pragma unroll
    for (int t = 0; t < 8; ++t) {
      v16bf b = load_b(&Bs[it & 1][16 * t][0], 40);
      acc[t] = wmma_bf16(a, b, acc[t]);
    }
    __syncthreads();  // all waves done reading buf before it is re-staged
  }

  const int lane = threadIdx.x & 31;
  const int n = lane & 15;
  const int mh = (lane >> 4) << 3;
#pragma unroll
  for (int t = 0; t < 8; ++t) {
    const int col = col0 + 16 * t + n;
    const float bv = bias[col];
#pragma unroll
    for (int r = 0; r < 8; ++r)
      C[(size_t)(row0 + mh + r) * Ndim + col] = (OutT)(acc[t][r] + bv);
  }
}

// ---------------------------------------------------------------------------
// Flash attention: grid (SEQ/128, BATCH*HEADS); block 256 = 8 waves, each wave
// owns 16 query rows; streams 64-key blocks with online softmax.
// K tile (shared by all 8 waves) staged via async DMA; V staged transposed.
// Q/K/V/Ctx layout: [BATCH*SEQ, DIM] bf16, head h = columns [h*128, +128).
// ---------------------------------------------------------------------------
__global__ __launch_bounds__(256) void attn_kernel(const bf16* __restrict__ Qb,
                                                   const bf16* __restrict__ Kb,
                                                   const bf16* __restrict__ Vb,
                                                   bf16* __restrict__ Ctx) {
  __shared__ alignas(16) bf16 Kt[64][128 + 8];   // K tile row-major (272B rows)
  __shared__ alignas(16) bf16 Vt[HDIM][64 + 8];  // V^T tile (144B rows)
  __shared__ alignas(16) bf16 Pl[8][16][64 + 8]; // per-wave probability tiles

  const int wave = threadIdx.x >> 5;
  const int lane = threadIdx.x & 31;
  const int n = lane & 15;
  const int mh = (lane >> 4) << 3;
  const int bb = blockIdx.y >> 4;
  const int h = blockIdx.y & 15;
  const int q0 = blockIdx.x * 128 + wave * 16;
  const size_t rowbase = (size_t)bb * SEQ;
  const int hcol = h * HDIM;

  // Q fragments for this wave's 16 rows, full head dim (4 x K=32)
  v16bf qf[4];
#pragma unroll
  for (int dc = 0; dc < 4; ++dc)
    qf[dc] = load_a(Qb + (rowbase + q0) * DIM + hcol + 32 * dc, DIM);

  float rowM[8], rowL[8];
  v8f cacc[8] = {};
#pragma unroll
  for (int r = 0; r < 8; ++r) { rowM[r] = -1e30f; rowL[r] = 0.f; }

  for (int jb = 0; jb < SEQ; jb += 64) {
    __syncthreads();  // prior tile reads (Kt/Vt) complete before overwrite
    // Stage K tile: 64 keys x 128 d, straight async copy (B operand reads rows)
#pragma unroll
    for (int i = 0; i < 2; ++i) {
      int idx = threadIdx.x + i * 256;  // 0..511 : 64 rows x 8 chunks of 16B
      int key = idx >> 3;
      int c = (idx & 7) * 8;
      async_copy_b128(Kb + (rowbase + jb + key) * DIM + hcol + c, &Kt[key][c]);
    }
    // Stage V^T tile (transpose in the LDS write; async DMA cannot transpose)
#pragma unroll
    for (int i = 0; i < 4; ++i) {
      int v = threadIdx.x + i * 256;
      int key = v >> 4;
      int dcol = (v & 15) << 3;
      bf16x8 d8 = *reinterpret_cast<const bf16x8*>(
          Vb + (rowbase + jb + key) * DIM + hcol + dcol);
#pragma unroll
      for (int e = 0; e < 8; ++e) Vt[dcol + e][key] = d8[e];
    }
    wait_asynccnt<0>();  // own async K chunks landed in LDS
    __syncthreads();     // all waves' K (async) + V (ds) writes visible

    // S = Q K^T : 16x64 per wave (4 N-tiles x 4 K-chunks), K from LDS
    v8f s[4] = {};
#pragma unroll
    for (int t = 0; t < 4; ++t) {
#pragma unroll
      for (int dc = 0; dc < 4; ++dc) {
        v16bf kf = load_b(&Kt[16 * t][32 * dc], 128 + 8);
        s[t] = wmma_bf16(qf[dc], kf, s[t]);
      }
    }
#pragma unroll
    for (int t = 0; t < 4; ++t)
#pragma unroll
      for (int r = 0; r < 8; ++r) s[t][r] *= SM_SCALE;

    // Online softmax: C-layout puts N across the 16-lane half, rows in VGPRs.
    float corr[8];
#pragma unroll
    for (int r = 0; r < 8; ++r) {
      float v = fmaxf(fmaxf(s[0][r], s[1][r]), fmaxf(s[2][r], s[3][r]));
      v = redmax16(v);
      float mnew = fmaxf(rowM[r], v);
      corr[r] = __expf(rowM[r] - mnew);
      rowM[r] = mnew;
      rowL[r] *= corr[r];
    }
#pragma unroll
    for (int dt = 0; dt < 8; ++dt)
#pragma unroll
      for (int r = 0; r < 8; ++r) cacc[dt][r] *= corr[r];
#pragma unroll
    for (int t = 0; t < 4; ++t)
#pragma unroll
      for (int r = 0; r < 8; ++r) s[t][r] = __expf(s[t][r] - rowM[r]);
#pragma unroll
    for (int r = 0; r < 8; ++r) {
      float sum = s[0][r] + s[1][r] + s[2][r] + s[3][r];
      rowL[r] += redsum16(sum);
    }

    // Re-layout P: C-layout -> A-operand via per-wave LDS tile (wave-private
    // region; compiler orders the round-trip via DScnt).
#pragma unroll
    for (int t = 0; t < 4; ++t)
#pragma unroll
      for (int r = 0; r < 8; ++r)
        Pl[wave][mh + r][16 * t + n] = (bf16)s[t][r];

    // ctx += P * V  (2 K-chunks of 32 keys x 8 d-tiles)
#pragma unroll
    for (int kt = 0; kt < 2; ++kt) {
      v16bf pf = load_a(&Pl[wave][0][kt * 32], 64 + 8);
#pragma unroll
      for (int dt = 0; dt < 8; ++dt) {
        v16bf vf = load_b(&Vt[16 * dt][kt * 32], 64 + 8);
        cacc[dt] = wmma_bf16(pf, vf, cacc[dt]);
      }
    }
  }

  // Normalize and store context (bf16, same [B*S, DIM] head-sliced layout)
#pragma unroll
  for (int r = 0; r < 8; ++r) rowL[r] = 1.f / rowL[r];
#pragma unroll
  for (int dt = 0; dt < 8; ++dt)
#pragma unroll
    for (int r = 0; r < 8; ++r)
      Ctx[(rowbase + q0 + mh + r) * DIM + hcol + 16 * dt + n] =
          (bf16)(cacc[dt][r] * rowL[r]);
}

// ---------------------------------------------------------------------------

extern "C" void kernel_launch(void* const* d_in, const int* in_sizes, int n_in,
                              void* d_out, int out_size, void* d_ws, size_t ws_size,
                              hipStream_t stream) {
  (void)in_sizes; (void)n_in; (void)out_size; (void)ws_size;
  const float* x  = (const float*)d_in[0];
  const float* wq = (const float*)d_in[1]; const float* bq = (const float*)d_in[2];
  const float* wk = (const float*)d_in[3]; const float* bk = (const float*)d_in[4];
  const float* wv = (const float*)d_in[5]; const float* bv = (const float*)d_in[6];
  const float* wo = (const float*)d_in[7]; const float* bo = (const float*)d_in[8];
  float* out = (float*)d_out;

  char* ws = (char*)d_ws;
  size_t off = 0;
  auto alloc = [&](size_t bytes) -> void* {
    void* p = ws + off;
    off += (bytes + 255) & ~(size_t)255;
    return p;
  };
  bf16* xb   = (bf16*)alloc((size_t)MROWS * DIM * sizeof(bf16));
  bf16* Wtq  = (bf16*)alloc((size_t)DIM * DIM * sizeof(bf16));
  bf16* Wtk  = (bf16*)alloc((size_t)DIM * DIM * sizeof(bf16));
  bf16* Wtv  = (bf16*)alloc((size_t)DIM * DIM * sizeof(bf16));
  bf16* Wto  = (bf16*)alloc((size_t)DIM * DIM * sizeof(bf16));
  bf16* Qb   = (bf16*)alloc((size_t)MROWS * DIM * sizeof(bf16));
  bf16* Kb   = (bf16*)alloc((size_t)MROWS * DIM * sizeof(bf16));
  bf16* Vb   = (bf16*)alloc((size_t)MROWS * DIM * sizeof(bf16));
  bf16* ctxb = (bf16*)alloc((size_t)MROWS * DIM * sizeof(bf16));

  // 1. x -> bf16
  {
    int nel = MROWS * DIM;
    cvt_f32_bf16<<<nel / 4 / 256, 256, 0, stream>>>(x, xb, nel);
  }
  // 2. weight transposes (f32 [K,N] -> bf16 [N,K])
  {
    dim3 g(DIM / 32, DIM / 32);
    transpose_f32_bf16<<<g, 256, 0, stream>>>(wq, Wtq, DIM);
    transpose_f32_bf16<<<g, 256, 0, stream>>>(wk, Wtk, DIM);
    transpose_f32_bf16<<<g, 256, 0, stream>>>(wv, Wtv, DIM);
    transpose_f32_bf16<<<g, 256, 0, stream>>>(wo, Wto, DIM);
  }
  // 3. QKV projections (bf16 out)
  {
    dim3 g(DIM / 128, MROWS / 128);
    gemm_k<bf16><<<g, 256, 0, stream>>>(xb, Wtq, bq, Qb, DIM, DIM);
    gemm_k<bf16><<<g, 256, 0, stream>>>(xb, Wtk, bk, Kb, DIM, DIM);
    gemm_k<bf16><<<g, 256, 0, stream>>>(xb, Wtv, bv, Vb, DIM, DIM);
  }
  // 4. flash attention
  {
    dim3 g(SEQ / 128, BATCH * 16);
    attn_kernel<<<g, 256, 0, stream>>>(Qb, Kb, Vb, ctxb);
  }
  // 5. output projection (f32 out)
  {
    dim3 g(DIM / 128, MROWS / 128);
    gemm_k<float><<<g, 256, 0, stream>>>(ctxb, Wto, bo, out, DIM, DIM);
  }
}
